// MultiHeadAttention_42365557408365
// MI455X (gfx1250) — matrix operand order repaired and checked by reading, not compiled, not run on hardware
//
#include <hip/hip_runtime.h>
#include <hip/hip_bf16.h>

typedef __bf16 bf16_t;
typedef __bf16 v16bf __attribute__((ext_vector_type(16)));
typedef __bf16 v8bf  __attribute__((ext_vector_type(8)));
typedef float  v8f   __attribute__((ext_vector_type(8)));

union Frag {
    v16bf v;
    v8bf  h[2];
};

#define WMMA_BF16(a, b, c) \
    __builtin_amdgcn_wmma_f32_16x16x32_bf16(false, (a), false, (b), (short)0, (c), false, false)

// Async copy 16B global -> LDS (gfx1250 GLOBAL_LOAD_ASYNC_TO_LDS_B128, ASYNCcnt).
// Generic shared pointers on amdgcn carry the LDS byte offset in the low 32 bits.
__device__ __forceinline__ void async_copy_b128(void* lds, const void* gptr) {
    unsigned ldsa = (unsigned)(unsigned long long)lds;
    asm volatile("global_load_async_to_lds_b128 %0, %1, off"
                 :: "v"(ldsa), "v"(gptr)
                 : "memory");
}
__device__ __forceinline__ void wait_async() {
    asm volatile("s_wait_asynccnt 0" ::: "memory");
}

// ---------------------------------------------------------------------------
// fp32 -> bf16 cast (elementwise; used for x)
// ---------------------------------------------------------------------------
__global__ void mha_cast_bf16(const float* __restrict__ in, bf16_t* __restrict__ out, int n) {
    int i = blockIdx.x * blockDim.x + threadIdx.x;
    if (i < n) out[i] = (bf16_t)in[i];
}

// ---------------------------------------------------------------------------
// fp32 [K][N] -> bf16 [N][K] cast + transpose (LDS 32x32 tiles, one-time)
// ---------------------------------------------------------------------------
__global__ void mha_cast_transpose(const float* __restrict__ in, bf16_t* __restrict__ out,
                                   int K, int N) {
    __shared__ float tile[32][33];
    const int k0 = blockIdx.x * 32;
    const int n0 = blockIdx.y * 32;
    const int tx = threadIdx.x & 31;
    const int ty = threadIdx.x >> 5;  // 0..7
#pragma unroll
    for (int i = ty; i < 32; i += 8)
        tile[i][tx] = in[(size_t)(k0 + i) * N + n0 + tx];
    __syncthreads();
#pragma unroll
    for (int i = ty; i < 32; i += 8)
        out[(size_t)(n0 + i) * K + k0 + tx] = (bf16_t)tile[tx][i];
}

// ---------------------------------------------------------------------------
// bf16 GEMM: out[M,N] = A[M,K] @ Bt[N,K]^T + bias[N]
// A row-major [M][K], Bt pre-transposed [N][K].
// Block tile 128x128, 8 waves, wave tile 32x64, K-step 32,
// double-buffered async LDS staging.
// ---------------------------------------------------------------------------
template <bool OUT_BF16>
__global__ void mha_gemm_bf16(const bf16_t* __restrict__ A, const bf16_t* __restrict__ Bt,
                              const float* __restrict__ bias, void* __restrict__ outp,
                              int M, int N, int K) {
    __shared__ bf16_t As[2][128][32];  // [buf][m][k]
    __shared__ bf16_t Bs[2][128][32];  // [buf][n][k]

    const int t    = threadIdx.x;
    const int lane = t & 31;
    const int w    = t >> 5;
    const int ln   = lane & 15;
    const int hi   = lane >> 4;
    const int wm   = w & 3;
    const int wn   = w >> 2;
    const int m0   = blockIdx.x * 128;
    const int n0   = blockIdx.y * 128;

    const bf16_t* Abase = A + (size_t)m0 * K;
    const bf16_t* Bbase = Bt + (size_t)n0 * K;

    v8f c[2][4];
#pragma unroll
    for (int mi = 0; mi < 2; ++mi)
#pragma unroll
        for (int ni = 0; ni < 4; ++ni)
            c[mi][ni] = (v8f){0.f, 0.f, 0.f, 0.f, 0.f, 0.f, 0.f, 0.f};

    const int nsteps = K >> 5;

    // Stage step 0 into buffer 0 (async, no VGPR round trip)
    {
        for (int ch = t; ch < 512; ch += 256) {
            int row = ch >> 2, eo = (ch & 3) * 8;
            async_copy_b128(&As[0][row][eo], Abase + (size_t)row * K + eo);
            async_copy_b128(&Bs[0][row][eo], Bbase + (size_t)row * K + eo);
        }
    }

    for (int ks = 0; ks < nsteps; ++ks) {
        wait_async();
        __syncthreads();
        const int cur = ks & 1;

        if (ks + 1 < nsteps) {
            const int kk = (ks + 1) << 5;
            const int nxt = cur ^ 1;
            for (int ch = t; ch < 512; ch += 256) {
                int row = ch >> 2, eo = (ch & 3) * 8;
                async_copy_b128(&As[nxt][row][eo], Abase + (size_t)row * K + kk + eo);
                async_copy_b128(&Bs[nxt][row][eo], Bbase + (size_t)row * K + kk + eo);
            }
        }

        // A fragments (ISA 16-bit A layout)
        Frag af[2];
        {
            const int ko = hi ? 8 : 0;
#pragma unroll
            for (int mi = 0; mi < 2; ++mi) {
                int row = wm * 32 + mi * 16 + ln;
                af[mi].h[0] = *(const v8bf*)&As[cur][row][ko];
                af[mi].h[1] = *(const v8bf*)&As[cur][row][ko + 16];
            }
        }
        // B fragments (ISA 16-bit B layout; Bs is [N][K])
        Frag bf[4];
        {
            const int kh = hi ? 8 : 0;
#pragma unroll
            for (int ni = 0; ni < 4; ++ni) {
                int col = wn * 64 + ni * 16 + ln;
                bf[ni].h[0] = *(const v8bf*)&Bs[cur][col][kh];
                bf[ni].h[1] = *(const v8bf*)&Bs[cur][col][kh + 16];
            }
        }
#pragma unroll
        for (int mi = 0; mi < 2; ++mi)
#pragma unroll
            for (int ni = 0; ni < 4; ++ni)
                c[mi][ni] = WMMA_BF16(af[mi].v, bf[ni].v, c[mi][ni]);
    }

    // Store (C/D layout: row = r + 8*hi, col = lane&15)
#pragma unroll
    for (int mi = 0; mi < 2; ++mi)
#pragma unroll
        for (int ni = 0; ni < 4; ++ni)
#pragma unroll
            for (int r = 0; r < 8; ++r) {
                int rowg = m0 + wm * 32 + mi * 16 + r + 8 * hi;
                int colg = n0 + wn * 64 + ni * 16 + ln;
                float v = c[mi][ni][r] + bias[colg];
                if (OUT_BF16)
                    ((bf16_t*)outp)[(size_t)rowg * N + colg] = (bf16_t)v;
                else
                    ((float*)outp)[(size_t)rowg * N + colg] = v;
            }
}

// ---------------------------------------------------------------------------
// Flash attention (causal). qkv: [B,S,3E] bf16; ctx: [B,S,E] bf16.
// One block per (b, h, 128-query tile); wave owns 16 query rows.
// K tile staged via async-to-LDS; V transposed during staging. Double-buffered.
// ---------------------------------------------------------------------------
__global__ void mha_flash_attn(const bf16_t* __restrict__ qkv, bf16_t* __restrict__ ctx) {
    const int S_ = 2048, E_ = 1024, TE = 3072;
    __shared__ bf16_t Kt[2][32][64];    // [buf][key][d]
    __shared__ bf16_t Vt[2][64][32];    // [buf][d][key]
    __shared__ bf16_t Pb[8][16][32];    // per-wave P transpose buffer

    const int t    = threadIdx.x;
    const int lane = t & 31;
    const int w    = t >> 5;
    const int ln   = lane & 15;
    const int hi   = lane >> 4;

    const int blk = blockIdx.x;
    const int qt  = blk & 15;
    const int h   = (blk >> 4) & 15;
    const int b   = blk >> 8;
    const int q0  = qt * 128;

    // Q fragments: rows q0 + w*16 + ln, two d-chunks of 32
    Frag qf[2];
    {
        const int ko   = hi ? 8 : 0;
        const int qrow = q0 + w * 16 + ln;
        const bf16_t* qb = qkv + (size_t)(b * S_ + qrow) * TE + h * 64;
#pragma unroll
        for (int kc = 0; kc < 2; ++kc) {
            const bf16_t* p = qb + kc * 32 + ko;
            qf[kc].h[0] = *(const v8bf*)p;
            qf[kc].h[1] = *(const v8bf*)(p + 16);
        }
    }

    float m[8], l[8];
    v8f acc[4];
#pragma unroll
    for (int r = 0; r < 8; ++r) { m[r] = -1e30f; l[r] = 0.f; }
#pragma unroll
    for (int dt = 0; dt < 4; ++dt)
        acc[dt] = (v8f){0.f, 0.f, 0.f, 0.f, 0.f, 0.f, 0.f, 0.f};

    const int key  = t >> 3;
    const int dc   = (t & 7) * 8;
    const int nkb  = (q0 + 128) >> 5;

    // Stage kb = 0 into buffer 0
    {
        size_t rowoff = (size_t)(b * S_ + key) * TE + h * 64 + dc;
        async_copy_b128(&Kt[0][key][dc], qkv + rowoff + E_);
        uint4 vv = *(const uint4*)(qkv + rowoff + 2 * E_);
        const bf16_t* ve = (const bf16_t*)&vv;
#pragma unroll
        for (int i = 0; i < 8; ++i) Vt[0][dc + i][key] = ve[i];
    }

    for (int kb = 0; kb < nkb; ++kb) {
        wait_async();
        __syncthreads();
        const int cur = kb & 1;

        if (kb + 1 < nkb) {
            const int nxt = cur ^ 1;
            size_t rowoff = (size_t)(b * S_ + (kb + 1) * 32 + key) * TE + h * 64 + dc;
            async_copy_b128(&Kt[nxt][key][dc], qkv + rowoff + E_);
            uint4 vv = *(const uint4*)(qkv + rowoff + 2 * E_);
            const bf16_t* ve = (const bf16_t*)&vv;
#pragma unroll
            for (int i = 0; i < 8; ++i) Vt[nxt][dc + i][key] = ve[i];
        }

        // Scores: S = Q @ K^T, two 16-key column tiles
        v8f s[2];
#pragma unroll
        for (int nt = 0; nt < 2; ++nt) {
            v8f cc = (v8f){0.f, 0.f, 0.f, 0.f, 0.f, 0.f, 0.f, 0.f};
            const int col = nt * 16 + ln;
            const int kh  = hi ? 8 : 0;
#pragma unroll
            for (int kc = 0; kc < 2; ++kc) {
                Frag bfr;
                const bf16_t* p = &Kt[cur][col][kc * 32 + kh];
                bfr.h[0] = *(const v8bf*)p;
                bfr.h[1] = *(const v8bf*)(p + 16);
                cc = WMMA_BF16(qf[kc].v, bfr.v, cc);
            }
            s[nt] = cc;
        }

        // Scale + causal mask
        const float scale = 0.125f;  // 1/sqrt(64)
#pragma unroll
        for (int nt = 0; nt < 2; ++nt)
#pragma unroll
            for (int r = 0; r < 8; ++r) {
                int key_abs = kb * 32 + nt * 16 + ln;
                int q_abs   = q0 + w * 16 + r + 8 * hi;
                float v = s[nt][r] * scale;
                s[nt][r] = (key_abs > q_abs) ? -1e30f : v;
            }

        // Online softmax (row = 16-lane group at fixed r)
        float alpha[8];
#pragma unroll
        for (int r = 0; r < 8; ++r) {
            float mx = fmaxf(s[0][r], s[1][r]);
#pragma unroll
            for (int msk = 1; msk < 16; msk <<= 1)
                mx = fmaxf(mx, __shfl_xor(mx, msk, 32));
            float mn = fmaxf(m[r], mx);
            float al = __expf(m[r] - mn);
            float p0 = __expf(s[0][r] - mn);
            float p1 = __expf(s[1][r] - mn);
            s[0][r] = p0;
            s[1][r] = p1;
            float rs = p0 + p1;
#pragma unroll
            for (int msk = 1; msk < 16; msk <<= 1)
                rs += __shfl_xor(rs, msk, 32);
            l[r]     = l[r] * al + rs;
            m[r]     = mn;
            alpha[r] = al;
        }
#pragma unroll
        for (int dt = 0; dt < 4; ++dt)
#pragma unroll
            for (int r = 0; r < 8; ++r) acc[dt][r] *= alpha[r];

        // Transpose P through per-wave LDS into an A fragment
#pragma unroll
        for (int nt = 0; nt < 2; ++nt)
#pragma unroll
            for (int r = 0; r < 8; ++r)
                Pb[w][r + 8 * hi][nt * 16 + ln] = (bf16_t)s[nt][r];

        Frag pf;
        {
            const int ko = hi ? 8 : 0;
            pf.h[0] = *(const v8bf*)&Pb[w][ln][ko];
            pf.h[1] = *(const v8bf*)&Pb[w][ln][ko + 16];
        }

        // acc += P @ V
        {
            const int kh = hi ? 8 : 0;
#pragma unroll
            for (int dt = 0; dt < 4; ++dt) {
                Frag vf;
                const bf16_t* p = &Vt[cur][dt * 16 + ln][kh];
                vf.h[0] = *(const v8bf*)p;
                vf.h[1] = *(const v8bf*)(p + 16);
                acc[dt] = WMMA_BF16(pf.v, vf.v, acc[dt]);
            }
        }
    }

    // Epilogue: normalize, store ctx[b, s, h*64 + d] (bf16)
#pragma unroll
    for (int dt = 0; dt < 4; ++dt)
#pragma unroll
        for (int r = 0; r < 8; ++r) {
            int srow = q0 + w * 16 + r + 8 * hi;
            int col  = h * 64 + dt * 16 + ln;
            float ov = acc[dt][r] / l[r];
            ctx[(size_t)(b * S_ + srow) * E_ + col] = (bf16_t)ov;
        }
}

// ---------------------------------------------------------------------------
// Host launcher
// ---------------------------------------------------------------------------
extern "C" void kernel_launch(void* const* d_in, const int* in_sizes, int n_in,
                              void* d_out, int out_size, void* d_ws, size_t ws_size,
                              hipStream_t stream) {
    const int B = 2, S = 2048, E = 1024;
    const int M  = B * S;      // 4096
    const int TE = 3 * E;      // 3072

    const float* x     = (const float*)d_in[0];
    const float* W_qkv = (const float*)d_in[1];
    const float* b_qkv = (const float*)d_in[2];
    const float* W_out = (const float*)d_in[3];
    const float* b_out = (const float*)d_in[4];
    float* out = (float*)d_out;

    const size_t n_x    = (size_t)M * E;    // 4194304
    const size_t n_wqkv = (size_t)E * TE;   // 3145728
    const size_t n_wout = (size_t)E * E;    // 1048576
    const size_t n_qkv  = (size_t)M * TE;   // 12582912

    bf16_t* xb    = (bf16_t*)d_ws;
    bf16_t* wqkvt = xb + n_x;        // [3E][E] (transposed)
    bf16_t* woutt = wqkvt + n_wqkv;  // [E][E]  (transposed)
    bf16_t* qkvb  = woutt + n_wout;
    bf16_t* ctxb  = qkvb + n_qkv;

    // 1) cast x; cast+transpose weights (one-time, ~25 MB traffic)
    mha_cast_bf16<<<(int)((n_x + 255) / 256), 256, 0, stream>>>(x, xb, (int)n_x);
    mha_cast_transpose<<<dim3(E / 32, TE / 32), 256, 0, stream>>>(W_qkv, wqkvt, E, TE);
    mha_cast_transpose<<<dim3(E / 32, E / 32), 256, 0, stream>>>(W_out, woutt, E, E);

    // 2) qkv = x @ W_qkv + b_qkv  (bf16 out)
    mha_gemm_bf16<true><<<dim3(M / 128, TE / 128), 256, 0, stream>>>(
        xb, wqkvt, b_qkv, (void*)qkvb, M, TE, E);

    // 3) flash attention -> ctx (bf16)
    mha_flash_attn<<<B * 16 * (S / 128), 256, 0, stream>>>(qkvb, ctxb);

    // 4) out = ctx @ W_out + b_out  (fp32 out)
    mha_gemm_bf16<false><<<dim3(M / 128, E / 128), 256, 0, stream>>>(
        ctxb, woutt, b_out, (void*)out, M, E, E);
}
